// MyGCN_4483945857509
// MI455X (gfx1250) — compile-verified, matrix operand order checked by reading, not run on hardware
//
#include <hip/hip_runtime.h>
#include <hip/hip_bf16.h>

typedef __attribute__((ext_vector_type(16))) _Float16 v16h;
typedef __attribute__((ext_vector_type(8)))  float    v8f;

#define F_DIM 128
// Pre-swizzled W image: [8 col tiles][4 k-steps][32 lanes][16 halves]
#define WF_ELEMS (8 * 4 * 32 * 16)   // 16384 halves = 32 KB

// ---------------------------------------------------------------------------
// Pre-swizzle W (fp32 row-major 128x128) into the exact f16 B-fragment layout
// for v_wmma_f32_16x16x32_f16. Runs once per layer; 16384 threads.
// B-frag map: lane gives N = lane&15 (plus col tile), K = ks*32 + kOff(lane)
//             + e + (e<8?0:8), kOff = (lane<16)?0:8.
// ---------------------------------------------------------------------------
__global__ void k_wswz(const float* __restrict__ W, _Float16* __restrict__ Wf) {
  int t = blockIdx.x * blockDim.x + threadIdx.x;
  if (t >= WF_ELEMS) return;
  int e    = t & 15;
  int lane = (t >> 4) & 31;
  int ks   = (t >> 9) & 3;
  int c    = t >> 11;                       // col tile 0..7
  int kOff = (lane < 16) ? 0 : 8;
  int k    = ks * 32 + kOff + e + ((e < 8) ? 0 : 8);
  int col  = c * 16 + (lane & 15);
  Wf[t] = (_Float16)W[k * F_DIM + col];
}

// ---------------------------------------------------------------------------
// GEMM: C[N,128] = A[N,128] (fp32, cvt->f16) x W (pre-swizzled f16 frags)
// One block = 256 threads = 8 waves; block covers 16 rows, wave w covers
// column tile w. Per wave: 4x v_wmma_f32_16x16x32_f16 over K=128.
// B fragment = single 32B vector load per k-step (contiguous 1KB per wave).
// ---------------------------------------------------------------------------
__global__ __launch_bounds__(256) void gcn_gemm_f16wmma(
    const float* __restrict__ A, const _Float16* __restrict__ Wf,
    float* __restrict__ C)
{
  const int lane = threadIdx.x & 31;
  const int wave = threadIdx.x >> 5;          // column tile 0..7
  const int m15  = lane & 15;
  const int row  = blockIdx.x * 16 + m15;     // A row for this lane
  const int col  = wave * 16 + m15;           // C column for this lane
  const int kOff = (lane < 16) ? 0 : 8;       // K offset per half-wave

  const _Float16* wfrag = Wf + (size_t)wave * 2048 + (size_t)lane * 16;

  v8f acc = {};
  #pragma unroll
  for (int ks = 0; ks < 4; ++ks) {
    const int kb = ks * 32;
    v16h af;
    #pragma unroll
    for (int e = 0; e < 16; e += 2) {
      const int k = kb + kOff + e + ((e < 8) ? 0 : 8);   // ISA A-frag K map
      const float2 av = *(const float2*)&A[(size_t)row * F_DIM + k];
      af[e]     = (_Float16)av.x;
      af[e + 1] = (_Float16)av.y;
    }
    const v16h bf = *(const v16h*)(wfrag + ks * 512);
    acc = __builtin_amdgcn_wmma_f32_16x16x32_f16(
        /*neg_a=*/false, af, /*neg_b=*/false, bf,
        /*c_mod=*/(short)0, acc, /*reuse_a=*/false, /*reuse_b=*/false);
  }
  const int mBase = blockIdx.x * 16 + ((lane < 16) ? 0 : 8);
  #pragma unroll
  for (int r = 0; r < 8; ++r)
    C[(size_t)(mBase + r) * F_DIM + col] = acc[r];
}

// ---------------------------------------------------------------------------
// Small helper kernels (degree/norm, scatter aggregation, bias+relu, pooling)
// ---------------------------------------------------------------------------
__global__ void k_fill(float* __restrict__ p, float v, int n) {
  int i = blockIdx.x * blockDim.x + threadIdx.x;
  if (i < n) p[i] = v;
}

__global__ void k_deg(const int* __restrict__ dst, float* __restrict__ deg, int E) {
  int e = blockIdx.x * blockDim.x + threadIdx.x;
  if (e < E) atomicAdd(&deg[dst[e]], 1.0f);
}

__global__ void k_dinv(float* __restrict__ deg, int n) {
  int i = blockIdx.x * blockDim.x + threadIdx.x;
  if (i < n) {
    float d = deg[i];
    deg[i] = (d > 0.0f) ? rsqrtf(d) : 0.0f;   // in-place deg -> deg^-1/2
  }
}

// agg[i,:] = lin[i,:] * dinv[i]^2   (self-loop term; also initializes agg)
__global__ void k_selfinit(const float* __restrict__ lin,
                           const float* __restrict__ dinv,
                           float* __restrict__ agg, int n) {
  int idx = blockIdx.x * blockDim.x + threadIdx.x;       // n*32 threads
  int i = idx >> 5;
  if (i >= n) return;
  int f = (idx & 31) << 2;
  float di = dinv[i];
  float w = di * di;
  const float4 v = *(const float4*)&lin[(size_t)i * F_DIM + f];
  float4 r; r.x = v.x * w; r.y = v.y * w; r.z = v.z * w; r.w = v.w * w;
  *(float4*)&agg[(size_t)i * F_DIM + f] = r;
}

// agg[dst,:] += lin[src,:] * dinv[src]*dinv[dst]   (L2-resident atomics)
__global__ void k_edges(const int* __restrict__ src, const int* __restrict__ dst,
                        const float* __restrict__ dinv,
                        const float* __restrict__ lin,
                        float* __restrict__ agg, int E) {
  int idx = blockIdx.x * blockDim.x + threadIdx.x;       // E*32 threads
  int e = idx >> 5;
  if (e >= E) return;
  int f = (idx & 31) << 2;
  int s = src[e], d = dst[e];
  float w = dinv[s] * dinv[d];
  const float4 v = *(const float4*)&lin[(size_t)s * F_DIM + f];
  float* ap = &agg[(size_t)d * F_DIM + f];
  atomicAdd(ap + 0, v.x * w);
  atomicAdd(ap + 1, v.y * w);
  atomicAdd(ap + 2, v.z * w);
  atomicAdd(ap + 3, v.w * w);
}

__global__ void k_biasrelu(const float* __restrict__ agg,
                           const float* __restrict__ b,
                           float* __restrict__ act, int n /* = N*F */) {
  int idx = blockIdx.x * blockDim.x + threadIdx.x;
  if (idx >= n) return;
  int f = idx & (F_DIM - 1);
  act[idx] = fmaxf(agg[idx] + b[f], 0.0f);
}

__global__ void k_pool(const float* __restrict__ act, const int* __restrict__ batch,
                       float* __restrict__ pooled, float* __restrict__ cnt, int n) {
  int idx = blockIdx.x * blockDim.x + threadIdx.x;       // n*32 threads
  int i = idx >> 5;
  if (i >= n) return;
  int f = (idx & 31) << 2;
  int g = batch[i];
  const float4 v = *(const float4*)&act[(size_t)i * F_DIM + f];
  float* pp = &pooled[(size_t)g * F_DIM + f];
  atomicAdd(pp + 0, v.x);
  atomicAdd(pp + 1, v.y);
  atomicAdd(pp + 2, v.z);
  atomicAdd(pp + 3, v.w);
  if ((idx & 31) == 0) atomicAdd(&cnt[g], 1.0f);
}

__global__ void k_final(const float* __restrict__ pooled, const float* __restrict__ cnt,
                        const float* __restrict__ lw, const float* __restrict__ lb,
                        float* __restrict__ out, int G) {
  int g = blockIdx.x * blockDim.x + threadIdx.x;
  if (g >= G) return;
  float s = 0.0f;
  #pragma unroll 4
  for (int f = 0; f < F_DIM; ++f) s += pooled[(size_t)g * F_DIM + f] * lw[f];
  out[g] = s / fmaxf(cnt[g], 1.0f) + lb[0];
}

// ---------------------------------------------------------------------------
extern "C" void kernel_launch(void* const* d_in, const int* in_sizes, int n_in,
                              void* d_out, int out_size, void* d_ws, size_t ws_size,
                              hipStream_t stream) {
  const int N = in_sizes[0] / F_DIM;     // 40000
  const int E = in_sizes[1] / 2;         // 600000
  const int G = out_size;                // 64

  const float* x     = (const float*)d_in[0];
  const int*   ei    = (const int*)d_in[1];
  const int*   batch = (const int*)d_in[2];
  const float* Wm[3] = {(const float*)d_in[3], (const float*)d_in[5], (const float*)d_in[7]};
  const float* bv[3] = {(const float*)d_in[4], (const float*)d_in[6], (const float*)d_in[8]};
  const float* lw    = (const float*)d_in[9];
  const float* lb    = (const float*)d_in[10];
  float*       out   = (float*)d_out;
  const int*   src   = ei;
  const int*   dst   = ei + E;

  // Workspace carve-up (fp32 units). Whole live set ~62 MB -> L2-resident.
  float* ws = (float*)d_ws;
  size_t o = 0;
  float*     dinv   = ws + o; o += ((size_t)N + 63) & ~(size_t)63;
  float*     lin    = ws + o; o += (size_t)N * F_DIM;
  float*     agg    = ws + o; o += (size_t)N * F_DIM;
  float*     act    = ws + o; o += (size_t)N * F_DIM;
  float*     pooled = ws + o; o += (size_t)G * F_DIM;
  float*     cnt    = ws + o; o += (size_t)G;        // contiguous after pooled
  o = (o + 63) & ~(size_t)63;                        // 32B+ align for v16h
  _Float16*  wf     = (_Float16*)(ws + o); o += WF_ELEMS / 2;
  if (o * sizeof(float) > ws_size) return;           // deterministic guard

  const int T = 256;
  auto cdiv = [](long long a, long long b) { return (int)((a + b - 1) / b); };

  // Degrees (self-loop => init 1.0), then deg^-1/2 in place.
  k_fill<<<cdiv(N, T), T, 0, stream>>>(dinv, 1.0f, N);
  k_deg <<<cdiv(E, T), T, 0, stream>>>(dst, dinv, E);
  k_dinv<<<cdiv(N, T), T, 0, stream>>>(dinv, N);

  // Three GCN layers: swizzle W -> WMMA GEMM -> self term -> edge scatter
  // -> bias+ReLU.
  const float* cur = x;
  for (int l = 0; l < 3; ++l) {
    k_wswz<<<cdiv(WF_ELEMS, T), T, 0, stream>>>(Wm[l], wf);
    gcn_gemm_f16wmma<<<N / 16, 256, 0, stream>>>(cur, wf, lin);
    k_selfinit<<<cdiv((long long)N * 32, T), T, 0, stream>>>(lin, dinv, agg, N);
    k_edges   <<<cdiv((long long)E * 32, T), T, 0, stream>>>(src, dst, dinv, lin, agg, E);
    k_biasrelu<<<cdiv((long long)N * F_DIM, T), T, 0, stream>>>(agg, bv[l], act, N * F_DIM);
    cur = act;
  }

  // Mean pool per graph + linear head.
  k_fill <<<cdiv(G * F_DIM + G, T), T, 0, stream>>>(pooled, 0.0f, G * F_DIM + G);
  k_pool <<<cdiv((long long)N * 32, T), T, 0, stream>>>(act, batch, pooled, cnt, N);
  k_final<<<1, 64, 0, stream>>>(pooled, cnt, lw, lb, out, G);
}
